// AttentionAnalyzer_50835232915810
// MI455X (gfx1250) — matrix operand order repaired and checked by reading, not compiled, not run on hardware
//
#include <hip/hip_runtime.h>

// ---------------------------------------------------------------------------
// AttentionAnalyzer on MI455X (gfx1250): bf16 WMMA GRU scan, persistent WGP.
// ---------------------------------------------------------------------------

typedef __bf16 bf16_t;
typedef __attribute__((ext_vector_type(16))) __bf16 v16bf;
typedef __attribute__((ext_vector_type(8)))  float  v8f;
typedef __attribute__((ext_vector_type(4)))  unsigned int u32x4;

#define S_LEN 256
#define BATCH 64
#define EMB   300
#define EPAD  320   // E padded to multiple of 32 (zero fill)
#define HID   256
#define G3    768   // 3*H (gates r,z,n stacked, torch order)
#define DD    512   // 2*H
#define NTOT  1536  // 2 directions * 3H
#define OUTC  2

// ---- WMMA fragment helpers (16x16x32 bf16; contiguous k order used for both A and B) -------------------
// A (16x32, MxK): lane L -> M = L&15; element i -> K = k0 + 16*(L>>4) + i
//                 -> two 16B loads
__device__ __forceinline__ v16bf ld_a(const bf16_t* p) {
  union { u32x4 q[2]; v16bf v; } u;
  u.q[0] = *reinterpret_cast<const u32x4*>(p);
  u.q[1] = *reinterpret_cast<const u32x4*>(p + 8);
  return u.v;
}
// B (32x16, KxN): lane L -> N = L&15; element i -> K = k0 + 16*(L>>4) + i
//                 -> one contiguous 32B load
__device__ __forceinline__ v16bf ld_b(const bf16_t* p) {
  return *reinterpret_cast<const v16bf*>(p);
}
__device__ __forceinline__ v8f wmma_bf(v16bf a, v16bf b, v8f c) {
  return __builtin_amdgcn_wmma_f32_16x16x32_bf16(false, a, false, b,
                                                 (short)0, c, false, false);
}
__device__ __forceinline__ float sigmoidf_(float x) {
  return 1.f / (1.f + __expf(-x));
}

// ---- Prep: gather embedding rows -> bf16, K padded 300->320 ----------------
__global__ __launch_bounds__(256)
void k_emb_gather(const int* __restrict__ X, const float* __restrict__ embd,
                  bf16_t* __restrict__ dst) {
  int idx = blockIdx.x * blockDim.x + threadIdx.x;
  if (idx >= S_LEN * BATCH * EPAD) return;
  int r = idx / EPAD, c = idx % EPAD;
  float v = (c < EMB) ? embd[(size_t)X[r] * EMB + c] : 0.f;
  dst[idx] = (bf16_t)v;
}

// ---- Prep: f32 weight -> bf16, optional zero-pad of K ----------------------
__global__ __launch_bounds__(256)
void k_cvt_pad(const float* __restrict__ src, bf16_t* __restrict__ dst,
               int rows, int scols, int dcols) {
  int idx = blockIdx.x * blockDim.x + threadIdx.x;
  if (idx >= rows * dcols) return;
  int r = idx / dcols, c = idx % dcols;
  dst[idx] = (bf16_t)((c < scols) ? src[(size_t)r * scols + c] : 0.f);
}

// ---- Phase 1: gi0[s*B+b][dir*768 + n] = emb @ w_ih_0{f,b}.T + b_ih --------
// 16384 x 1536 output, K=320.  One 16x16 tile per wave, 8 waves per block.
__global__ __launch_bounds__(256)
void k_gi0(const bf16_t* __restrict__ emb, const bf16_t* __restrict__ wih0,
           const float* __restrict__ bih0f, const float* __restrict__ bih0b,
           float* __restrict__ gi0) {
  const int lane = threadIdx.x & 31;
  const int wv   = threadIdx.x >> 5;
  const int tile = blockIdx.x * 8 + wv;               // 1024*96 tiles
  if (tile >= (S_LEN * BATCH / 16) * (NTOT / 16)) return;
  const int nt = tile % (NTOT / 16);
  const int mt = tile / (NTOT / 16);
  const int l15 = lane & 15, lhi = lane >> 4;
  const int n_g  = nt * 16 + l15;                     // 0..1535
  const int dir  = n_g / G3;
  const int n_in = n_g % G3;
  const bf16_t* ab = emb  + (size_t)(mt * 16 + l15) * EPAD + (lhi << 4);
  const bf16_t* bb = wih0 + (size_t)dir * G3 * EPAD
                          + (size_t)n_in * EPAD + (lhi << 4);
  const float bias = dir ? bih0b[n_in] : bih0f[n_in];
  v8f c;
#pragma unroll
  for (int j = 0; j < 8; ++j) c[j] = bias;
#pragma unroll
  for (int k0 = 0; k0 < EPAD; k0 += 32)
    c = wmma_bf(ld_a(ab + k0), ld_b(bb + k0), c);
  const int Mb = mt * 16 + (lhi << 3);
#pragma unroll
  for (int j = 0; j < 8; ++j)
    gi0[(size_t)(Mb + j) * NTOT + n_g] = c[j];
}

// ---- Phase 2: persistent single-WGP GRU scan -------------------------------
// Hidden state (h0f,h0b,h1f,h1b) double-buffered bf16 in LDS: 8*32KB = 256KB.
// 32 waves; each wave owns 4 gate-tiles per layer-phase and does the full
// (r,z,n) GRU update for its 16x16 h-tile in registers.
__global__ __launch_bounds__(1024)
void k_scan(const float* __restrict__ gi0,
            const bf16_t* __restrict__ whh0,  // [2][768][256] bf16
            const bf16_t* __restrict__ wih1,  // [2][768][512] bf16
            const bf16_t* __restrict__ whh1,  // [2][768][256] bf16
            const float* __restrict__ bhh0f, const float* __restrict__ bhh0b,
            const float* __restrict__ bih1f, const float* __restrict__ bhh1f,
            const float* __restrict__ bih1b, const float* __restrict__ bhh1b,
            float* __restrict__ yt) {
  extern __shared__ bf16_t hsm[];   // 8 buffers of [64][256] bf16
  const int tid  = threadIdx.x;
  const int lane = tid & 31, wv = tid >> 5;
  const int l15  = lane & 15, lhi = lane >> 4;
  const int HB   = BATCH * HID;     // 16384 elems per buffer

  for (int i = tid; i < 8 * HB; i += 1024) hsm[i] = (bf16_t)0.f;
  __syncthreads();

  for (int s = 0; s < S_LEN; ++s) {
    const int po = s & 1, pn = po ^ 1;

    // L2 warm-up for next step's pre-gates
    if (s + 1 < S_LEN)
      __builtin_prefetch(gi0 + (size_t)((s + 1) * BATCH + (wv << 1)) * NTOT, 0, 0);

    // ---------------- layer 0 (gh = h0_old @ w_hh.T; gi0 precomputed) ------
#pragma unroll
    for (int i = 0; i < 4; ++i) {
      const int t = wv * 4 + i;            // 0..127
      const int cell = t >> 6;             // 0=f, 1=b
      const int rem = t & 63;
      const int mt = rem >> 4, ct = rem & 15;
      const int n  = ct * 16 + l15;        // h column 0..255
      const bf16_t* hold = hsm + (size_t)(cell * 2 + po) * HB;
      bf16_t*       hnew = hsm + (size_t)(cell * 2 + pn) * HB;
      const bf16_t* w    = whh0 + (size_t)cell * G3 * HID;
      const float*  bhh  = cell ? bhh0b : bhh0f;
      const int Mb = mt * 16 + (lhi << 3);

      v8f cr, cz, cn; float gin[8];
      const float br = bhh[n], bz = bhh[HID + n], bn = bhh[2 * HID + n];
#pragma unroll
      for (int j = 0; j < 8; ++j) {
        const float* g = gi0 + (size_t)(s * BATCH + Mb + j) * NTOT
                             + (size_t)cell * G3;
        cr[j]  = g[n] + br;
        cz[j]  = g[HID + n] + bz;
        cn[j]  = bn;
        gin[j] = g[2 * HID + n];
      }
      const bf16_t* ar = hold + (size_t)(mt * 16 + l15) * HID + (lhi << 4);
      const bf16_t* wr = w + (size_t)(0 * HID + n) * HID + (lhi << 4);
      const bf16_t* wz = w + (size_t)(1 * HID + n) * HID + (lhi << 4);
      const bf16_t* wn = w + (size_t)(2 * HID + n) * HID + (lhi << 4);
#pragma unroll
      for (int k0 = 0; k0 < HID; k0 += 32) {
        v16bf a = ld_a(ar + k0);
        cr = wmma_bf(a, ld_b(wr + k0), cr);
        cz = wmma_bf(a, ld_b(wz + k0), cz);
        cn = wmma_bf(a, ld_b(wn + k0), cn);
      }
#pragma unroll
      for (int j = 0; j < 8; ++j) {
        float r  = sigmoidf_(cr[j]);
        float z  = sigmoidf_(cz[j]);
        float nn = tanhf(gin[j] + r * cn[j]);
        float ho = (float)hold[(size_t)(Mb + j) * HID + n];
        hnew[(size_t)(Mb + j) * HID + n] = (bf16_t)((1.f - z) * nn + z * ho);
      }
    }
    __syncthreads();

    // ---------------- layer 1 (x = y0 = [h0f_new | h0b_new], K=512;
    //                           gh over h1_old, K=256) ----------------------
#pragma unroll
    for (int i = 0; i < 4; ++i) {
      const int t = wv * 4 + i;
      const int cell = t >> 6;
      const int rem = t & 63;
      const int mt = rem >> 4, ct = rem & 15;
      const int n  = ct * 16 + l15;
      const bf16_t* h0fn = hsm + (size_t)(0 * 2 + pn) * HB;
      const bf16_t* h0bn = hsm + (size_t)(1 * 2 + pn) * HB;
      const bf16_t* h1o  = hsm + (size_t)((2 + cell) * 2 + po) * HB;
      bf16_t*       h1n  = hsm + (size_t)((2 + cell) * 2 + pn) * HB;
      const bf16_t* wi = wih1 + (size_t)cell * G3 * DD;
      const bf16_t* wh = whh1 + (size_t)cell * G3 * HID;
      const float* bi = cell ? bih1b : bih1f;
      const float* bh = cell ? bhh1b : bhh1f;
      const int Mb = mt * 16 + (lhi << 3);

      v8f cr, cz, cgi, cgh;
      const float vr = bi[n] + bh[n];
      const float vz = bi[HID + n] + bh[HID + n];
      const float vi = bi[2 * HID + n];
      const float vh = bh[2 * HID + n];
#pragma unroll
      for (int j = 0; j < 8; ++j) { cr[j] = vr; cz[j] = vz; cgi[j] = vi; cgh[j] = vh; }

      const int aoff = (mt * 16 + l15) * HID + (lhi << 4);
      const bf16_t* wir = wi + (size_t)(0 * HID + n) * DD + (lhi << 4);
      const bf16_t* wiz = wi + (size_t)(1 * HID + n) * DD + (lhi << 4);
      const bf16_t* win = wi + (size_t)(2 * HID + n) * DD + (lhi << 4);
#pragma unroll
      for (int k0 = 0; k0 < DD; k0 += 32) {       // input-gate part (y0)
        const bf16_t* asrc = (k0 < HID) ? h0fn : h0bn;
        v16bf a = ld_a(asrc + aoff + (k0 & (HID - 1)));
        cr  = wmma_bf(a, ld_b(wir + k0), cr);
        cz  = wmma_bf(a, ld_b(wiz + k0), cz);
        cgi = wmma_bf(a, ld_b(win + k0), cgi);
      }
      const bf16_t* whr = wh + (size_t)(0 * HID + n) * HID + (lhi << 4);
      const bf16_t* whz = wh + (size_t)(1 * HID + n) * HID + (lhi << 4);
      const bf16_t* whn = wh + (size_t)(2 * HID + n) * HID + (lhi << 4);
#pragma unroll
      for (int k0 = 0; k0 < HID; k0 += 32) {      // hidden-gate part
        v16bf a = ld_a(h1o + aoff + k0);
        cr  = wmma_bf(a, ld_b(whr + k0), cr);
        cz  = wmma_bf(a, ld_b(whz + k0), cz);
        cgh = wmma_bf(a, ld_b(whn + k0), cgh);
      }
#pragma unroll
      for (int j = 0; j < 8; ++j) {
        float r  = sigmoidf_(cr[j]);
        float z  = sigmoidf_(cz[j]);
        float nn = tanhf(cgi[j] + r * cgh[j]);
        float ho = (float)h1o[(size_t)(Mb + j) * HID + n];
        h1n[(size_t)(Mb + j) * HID + n] = (bf16_t)((1.f - z) * nn + z * ho);
      }
    }
    __syncthreads();
  }

  // final state lives in buffer index pn of step S-1 == 0 (S even)
  const bf16_t* h1f = hsm + (size_t)(2 * 2 + 0) * HB;
  const bf16_t* h1b = hsm + (size_t)(3 * 2 + 0) * HB;
  for (int i = tid; i < BATCH * DD; i += 1024) {
    int b = i >> 9, c = i & (DD - 1);
    yt[i] = (c < HID) ? (float)h1f[b * HID + c]
                      : (float)h1b[b * HID + (c - HID)];
  }
}

// ---- Phase 3: head.  alpha==1 (length-1 softmax) => ctx = yt@wo.T + bo,
//      logits = ctx@sw.T + sb, 2-class log_softmax.  One block per batch row.
__global__ __launch_bounds__(256)
void k_head(const float* __restrict__ yt, const float* __restrict__ wo,
            const float* __restrict__ bo, const float* __restrict__ sw,
            const float* __restrict__ sb, float* __restrict__ out) {
  __shared__ float ctx[DD];
  __shared__ float lg[2];
  const int b = blockIdx.x, tid = threadIdx.x;
  const float* y = yt + (size_t)b * DD;
  for (int n = tid; n < DD; n += 256) {
    const float* w = wo + (size_t)n * DD;
    float acc = bo[n];
    for (int k = 0; k < DD; ++k) acc += y[k] * w[k];
    ctx[n] = acc;
  }
  __syncthreads();
  if (tid < OUTC) {
    const float* w = sw + (size_t)tid * DD;
    float acc = sb[tid];
    for (int k = 0; k < DD; ++k) acc += ctx[k] * w[k];
    lg[tid] = acc;
  }
  __syncthreads();
  if (tid == 0) {
    float m = fmaxf(lg[0], lg[1]);
    float lse = m + __logf(__expf(lg[0] - m) + __expf(lg[1] - m));
    out[b * OUTC + 0] = lg[0] - lse;
    out[b * OUTC + 1] = lg[1] - lse;
  }
}

// ---------------------------------------------------------------------------
extern "C" void kernel_launch(void* const* d_in, const int* in_sizes, int n_in,
                              void* d_out, int out_size, void* d_ws, size_t ws_size,
                              hipStream_t stream) {
  (void)in_sizes; (void)n_in; (void)out_size; (void)ws_size;
  const int*   X       = (const int*)  d_in[0];
  const float* embd_w  = (const float*)d_in[1];
  const float* w_ih_0f = (const float*)d_in[2];
  const float* w_hh_0f = (const float*)d_in[3];
  const float* b_ih_0f = (const float*)d_in[4];
  const float* b_hh_0f = (const float*)d_in[5];
  const float* w_ih_0b = (const float*)d_in[6];
  const float* w_hh_0b = (const float*)d_in[7];
  const float* b_ih_0b = (const float*)d_in[8];
  const float* b_hh_0b = (const float*)d_in[9];
  const float* w_ih_1f = (const float*)d_in[10];
  const float* w_hh_1f = (const float*)d_in[11];
  const float* b_ih_1f = (const float*)d_in[12];
  const float* b_hh_1f = (const float*)d_in[13];
  const float* w_ih_1b = (const float*)d_in[14];
  const float* w_hh_1b = (const float*)d_in[15];
  const float* b_ih_1b = (const float*)d_in[16];
  const float* b_hh_1b = (const float*)d_in[17];
  // d_in[18..20] = attn_wq/wk/v: dead (softmax over length-1 sequence)
  const float* attn_wo = (const float*)d_in[21];
  const float* attn_bo = (const float*)d_in[22];
  const float* sent_w  = (const float*)d_in[23];
  const float* sent_b  = (const float*)d_in[24];
  float* out = (float*)d_out;

  // ---- workspace carving (bytes) ----
  char* p = (char*)d_ws;
  bf16_t* ws_emb  = (bf16_t*)p; p += (size_t)S_LEN * BATCH * EPAD * 2;   // 10.0 MB
  bf16_t* ws_wih0 = (bf16_t*)p; p += (size_t)2 * G3 * EPAD * 2;         // 0.94 MB
  bf16_t* ws_whh0 = (bf16_t*)p; p += (size_t)2 * G3 * HID * 2;          // 0.75 MB
  bf16_t* ws_wih1 = (bf16_t*)p; p += (size_t)2 * G3 * DD * 2;           // 1.5 MB
  bf16_t* ws_whh1 = (bf16_t*)p; p += (size_t)2 * G3 * HID * 2;          // 0.75 MB
  float*  ws_gi0  = (float*)p;  p += (size_t)S_LEN * BATCH * NTOT * 4;  // 96 MB
  float*  ws_yt   = (float*)p;  p += (size_t)BATCH * DD * 4;            // 128 KB

  // ---- phase 0: gather + bf16 conversion of operands ----
  {
    int tot = S_LEN * BATCH * EPAD;
    k_emb_gather<<<(tot + 255) / 256, 256, 0, stream>>>(X, embd_w, ws_emb);
  }
  auto cvt = [&](const float* src, bf16_t* dst, int rows, int sc, int dc) {
    int tot = rows * dc;
    k_cvt_pad<<<(tot + 255) / 256, 256, 0, stream>>>(src, dst, rows, sc, dc);
  };
  cvt(w_ih_0f, ws_wih0,             G3, EMB, EPAD);
  cvt(w_ih_0b, ws_wih0 + G3 * EPAD, G3, EMB, EPAD);
  cvt(w_hh_0f, ws_whh0,             G3, HID, HID);
  cvt(w_hh_0b, ws_whh0 + G3 * HID,  G3, HID, HID);
  cvt(w_ih_1f, ws_wih1,             G3, DD,  DD);
  cvt(w_ih_1b, ws_wih1 + G3 * DD,   G3, DD,  DD);
  cvt(w_hh_1f, ws_whh1,             G3, HID, HID);
  cvt(w_hh_1b, ws_whh1 + G3 * HID,  G3, HID, HID);

  // ---- phase 1: parallel layer-0 input pre-gates (bf16 WMMA GEMM) ----
  {
    int tiles = (S_LEN * BATCH / 16) * (NTOT / 16);   // 98304
    k_gi0<<<tiles / 8, 256, 0, stream>>>(ws_emb, ws_wih0, b_ih_0f, b_ih_0b, ws_gi0);
  }

  // ---- phase 2: persistent GRU scan, one WGP, 256KB dynamic LDS ----
  k_scan<<<1, 1024, 8 * BATCH * HID * (int)sizeof(bf16_t), stream>>>(
      ws_gi0, ws_whh0, ws_wih1, ws_whh1,
      b_hh_0f, b_hh_0b, b_ih_1f, b_hh_1f, b_ih_1b, b_hh_1b, ws_yt);

  // ---- phase 3: head ----
  k_head<<<BATCH, 256, 0, stream>>>(ws_yt, attn_wo, attn_bo, sent_w, sent_b, out);
}
